// CausalMultiHeadAttention_84121229459707
// MI455X (gfx1250) — matrix operand order repaired and checked by reading, not compiled, not run on hardware
//
#include <hip/hip_runtime.h>
#include <hip/hip_bf16.h>

typedef __attribute__((ext_vector_type(16))) _Float16 v16h;
typedef __attribute__((ext_vector_type(8)))  _Float16 v8h;
typedef __attribute__((ext_vector_type(8)))  float    v8f;
typedef __attribute__((ext_vector_type(4)))  int      v4i;

#define WMMA_F32_F16(a, b, c) \
  __builtin_amdgcn_wmma_f32_16x16x32_f16(false, (a), false, (b), (short)0, (c), false, false)

static constexpr int BATCH = 4;
static constexpr int T     = 2048;
static constexpr int C     = 768;
static constexpr int H     = 12;
static constexpr int DH    = 64;
static constexpr int N3    = 3 * C;          // 2304
static constexpr int MALL  = BATCH * T;      // 8192

union AFrag { v16h v; v8h h2[2]; };

typedef __attribute__((address_space(1))) v4i  gv4i;   // global int4
typedef __attribute__((address_space(3))) v4i  lv4i;   // LDS int4
typedef __attribute__((address_space(3))) void as3_void;

// 16-byte async copy: global -> LDS, tracked by ASYNCcnt (gfx1250).
__device__ __forceinline__ void async_cp16(const void* g, void* l) {
#if __has_builtin(__builtin_amdgcn_global_load_async_to_lds_b128)
  __builtin_amdgcn_global_load_async_to_lds_b128((gv4i*)g, (lv4i*)l, 0, 0);
#else
  unsigned lofs = (unsigned)(unsigned long long)(as3_void*)l;
  asm volatile("global_load_async_to_lds_b128 %0, %1, off"
               :: "v"(lofs), "v"((unsigned long long)g) : "memory");
#endif
}

template <int N>
__device__ __forceinline__ void wait_async() {
#if __has_builtin(__builtin_amdgcn_s_wait_asynccnt)
  __builtin_amdgcn_s_wait_asynccnt(N);
#else
  asm volatile("s_wait_asynccnt %0" :: "i"(N) : "memory");
#endif
}

// ---------------- prep kernels ----------------

__global__ void k_cvt_f16(const float* __restrict__ in, _Float16* __restrict__ out, int n) {
  int i = blockIdx.x * blockDim.x + threadIdx.x;
  if (i < n) out[i] = (_Float16)in[i];
}

// out[n][k] = (f16) in[k][n]   (in: [K][N] row-major, out: [N][K] row-major)
__global__ void k_transpose_f16(const float* __restrict__ in, _Float16* __restrict__ out,
                                int K, int N) {
  int i = blockIdx.x * blockDim.x + threadIdx.x;
  if (i >= K * N) return;
  int n = i / K, k = i - n * K;
  out[(size_t)n * K + k] = (_Float16)in[(size_t)k * N + n];
}

// ---------------- WMMA GEMM with async-LDS double-buffered staging ----------------
// A: [M][K] f16 row-major.  Bt: [N][K] f16 (i.e. B transposed).  D = A*B + bias.
// mode 0: N=2304 qkv epilogue -> scatter f16 Q [bh][t][d], K [bh][t][d], Vt [bh][d][t]
// mode 1: f32 output, out[m*N + n]
__global__ __launch_bounds__(256) void k_gemm(
    const _Float16* __restrict__ A, const _Float16* __restrict__ Bt,
    const float* __restrict__ bias, int K, int mode,
    _Float16* __restrict__ Qh, _Float16* __restrict__ Kh, _Float16* __restrict__ Vth,
    float* __restrict__ outf, int N) {
  __shared__ _Float16 As[2][128][32];   // [buf][m-in-block][k-in-step]
  __shared__ _Float16 Bs[2][128][32];   // [buf][n-in-block][k-in-step]

  const int tid   = threadIdx.x;
  const int lane  = tid & 31;
  const int wave  = tid >> 5;
  const bool hi   = lane >= 16;
  const int  ln   = lane & 15;
  const int waveM = wave & 3;    // 4 waves along M
  const int waveN = wave >> 2;   // 2 waves along N
  const int mbase = blockIdx.y * 128 + waveM * 32;
  const int nbase = blockIdx.x * 128 + waveN * 64;
  const int aoff  = hi ? 16 : 0;
  const int boff  = hi ? 16 : 0;

  const _Float16* Abase = A  + (size_t)blockIdx.y * 128 * K;
  const _Float16* Bbase = Bt + (size_t)blockIdx.x * 128 * K;

  // stage one 128x32 A tile + one 128x32 B tile (8 KB each); 4 async ops/thread
  auto issue = [&](int buf, int k0) {
#pragma unroll
    for (int t = 0; t < 2; t++) {
      const int c   = tid + t * 256;          // 512 16-byte chunks per tile
      const int row = c >> 2;
      const int off = (c & 3) * 8;            // halfword offset within 32-wide row
      async_cp16(Abase + (size_t)row * K + k0 + off, &As[buf][row][off]);
      async_cp16(Bbase + (size_t)row * K + k0 + off, &Bs[buf][row][off]);
    }
  };

  v8f acc[2][4] = {};
  issue(0, 0);
  const int S = K / 32;
  for (int s = 0; s < S; s++) {
    if (s + 1 < S) {
      issue((s + 1) & 1, (s + 1) * 32);
      wait_async<4>();    // newest 4 = next buffer; current buffer complete
    } else {
      wait_async<0>();
    }
    __syncthreads();      // every wave waited on its own portion -> tile resident

    const int cur = s & 1;
    AFrag a[2];
#pragma unroll
    for (int i = 0; i < 2; i++) {
      const _Float16* ar = &As[cur][waveM * 32 + i * 16 + ln][0];
      a[i].h2[0] = *(const v8h*)(ar + aoff);
      a[i].h2[1] = *(const v8h*)(ar + 8 + aoff);
    }
    v16h bf[4];
#pragma unroll
    for (int j = 0; j < 4; j++)
      bf[j] = *(const v16h*)(&Bs[cur][waveN * 64 + j * 16 + ln][0] + boff);
#pragma unroll
    for (int i = 0; i < 2; i++)
#pragma unroll
      for (int j = 0; j < 4; j++)
        acc[i][j] = WMMA_F32_F16(a[i].v, bf[j], acc[i][j]);
    __syncthreads();      // all reads of buffer `cur` done before it is refilled
  }

#pragma unroll
  for (int i = 0; i < 2; i++) {
#pragma unroll
    for (int j = 0; j < 4; j++) {
      const int ncol = nbase + j * 16 + ln;
      const float bv = bias[ncol];
#pragma unroll
      for (int r = 0; r < 8; r++) {
        const int m = mbase + i * 16 + r + (hi ? 8 : 0);
        const float val = acc[i][j][r] + bv;
        if (mode == 0) {
          const int b = m >> 11, t = m & (T - 1);
          const int sec = ncol / C;
          const int c = ncol - sec * C;
          const int h = c >> 6, d = c & 63;
          const size_t bh = (size_t)(b * H + h);
          const _Float16 v16 = (_Float16)val;
          if (sec == 0)      Qh [(bh * T + t) * DH + d] = v16;
          else if (sec == 1) Kh [(bh * T + t) * DH + d] = v16;
          else               Vth[(bh * DH + d) * T + t] = v16;
        } else {
          outf[(size_t)m * N + ncol] = val;
        }
      }
    }
  }
}

// ---------------- flash attention (one wave = one 16-row q tile) ----------------
__global__ __launch_bounds__(32) void k_flash(
    const _Float16* __restrict__ Qh, const _Float16* __restrict__ Kh,
    const _Float16* __restrict__ Vth, _Float16* __restrict__ Yh) {
  __shared__ _Float16 pb[16][32];
  const int lane = threadIdx.x & 31;
  const bool hi  = lane >= 16;
  const int  ln  = lane & 15;
  const int blk  = blockIdx.x;
  const int qt   = blk & (T / 16 - 1);     // 128 q-tiles
  const int bh   = blk >> 7;               // b*H + h
  const int h    = bh % H, b = bh / H;
  const int qbase = qt * 16;
  const float scale = 0.125f;              // 1/sqrt(64)
  const int aoff = hi ? 16 : 0;

  // Q A-fragments for the two Dh halves (K = 0..31 and 32..63)
  const _Float16* qrow = Qh + ((size_t)bh * T + qbase + ln) * DH;
  AFrag qa[2];
#pragma unroll
  for (int w = 0; w < 2; w++) {
    qa[w].h2[0] = *(const v8h*)(qrow + w * 32 + aoff);
    qa[w].h2[1] = *(const v8h*)(qrow + w * 32 + 8 + aoff);
  }

  float m_i[8], l_i[8];
#pragma unroll
  for (int r = 0; r < 8; r++) { m_i[r] = -1e30f; l_i[r] = 0.0f; }
  v8f acc[4] = {};

  const int kend = qbase + 16;             // causal: need columns 0 .. qbase+15
  for (int k0 = 0; k0 < kend; k0 += 32) {
    if (k0 + 32 < kend) {                  // prefetch next K/V tiles into caches
      __builtin_prefetch(Kh + ((size_t)bh * T + k0 + 32 + ln) * DH, 0, 3);
      __builtin_prefetch(Vth + ((size_t)bh * DH + ln) * T + k0 + 32, 0, 3);
    }
    // S = Q * K^T for 32 key columns (two 16x16 C tiles)
    v8f s[2];
#pragma unroll
    for (int ct = 0; ct < 2; ct++) {
      const _Float16* krow = Kh + ((size_t)bh * T + k0 + ct * 16 + ln) * DH;
      v16h kb0 = *(const v16h*)(krow + (hi ? 16 : 0));
      v16h kb1 = *(const v16h*)(krow + 32 + (hi ? 16 : 0));
      v8f z = {};
      z = WMMA_F32_F16(qa[0].v, kb0, z);
      z = WMMA_F32_F16(qa[1].v, kb1, z);
      s[ct] = z;
    }
    // online softmax per row (row M=r+8*hi lives across the 16 lanes of a half)
#pragma unroll
    for (int r = 0; r < 8; r++) {
      const int row = qbase + r + (hi ? 8 : 0);
      float s0 = s[0][r] * scale; if (k0 + ln > row)      s0 = -1e30f;
      float s1 = s[1][r] * scale; if (k0 + 16 + ln > row) s1 = -1e30f;
      float mt = fmaxf(s0, s1);
#pragma unroll
      for (int d = 1; d < 16; d <<= 1) mt = fmaxf(mt, __shfl_xor(mt, d, 16));
      const float mnew  = fmaxf(m_i[r], mt);
      const float alpha = __expf(m_i[r] - mnew);
      const float p0 = __expf(s0 - mnew);
      const float p1 = __expf(s1 - mnew);
      float rs = p0 + p1;
#pragma unroll
      for (int d = 1; d < 16; d <<= 1) rs += __shfl_xor(rs, d, 16);
      l_i[r] = l_i[r] * alpha + rs;
      m_i[r] = mnew;
#pragma unroll
      for (int j = 0; j < 4; j++) acc[j][r] *= alpha;
      pb[r + (hi ? 8 : 0)][ln]      = (_Float16)p0;
      pb[r + (hi ? 8 : 0)][16 + ln] = (_Float16)p1;
    }
    __syncthreads();   // single-wave block: S_NOP, just an ordering fence
    AFrag pf;
    pf.h2[0] = *(const v8h*)&pb[ln][aoff];
    pf.h2[1] = *(const v8h*)&pb[ln][8 + aoff];
    __syncthreads();
    // O += P (16x32) * V (32x64): Vt gives contiguous per-lane B fragments
#pragma unroll
    for (int j = 0; j < 4; j++) {
      const _Float16* vrow = Vth + ((size_t)bh * DH + j * 16 + ln) * T + k0 + (hi ? 16 : 0);
      v16h vb = *(const v16h*)vrow;
      acc[j] = WMMA_F32_F16(pf.v, vb, acc[j]);
    }
  }

  // write y (f16, row-major [B*T, C]); col = h*64 + d
#pragma unroll
  for (int j = 0; j < 4; j++) {
#pragma unroll
    for (int r = 0; r < 8; r++) {
      const int row = qbase + r + (hi ? 8 : 0);
      const int d = j * 16 + ln;
      Yh[((size_t)b * T + row) * C + h * DH + d] = (_Float16)(acc[j][r] / l_i[r]);
    }
  }
}

// ---------------- launch ----------------

extern "C" void kernel_launch(void* const* d_in, const int* in_sizes, int n_in,
                              void* d_out, int out_size, void* d_ws, size_t ws_size,
                              hipStream_t stream) {
  (void)in_sizes; (void)n_in; (void)out_size; (void)ws_size;
  const float* x      = (const float*)d_in[0];
  const float* w_attn = (const float*)d_in[1];
  const float* b_attn = (const float*)d_in[2];
  const float* w_proj = (const float*)d_in[3];
  const float* b_proj = (const float*)d_in[4];
  float* out = (float*)d_out;

  char* ws = (char*)d_ws;
  size_t off = 0;
  auto alloc16 = [&](size_t nelem) -> _Float16* {
    _Float16* p = (_Float16*)(ws + off);
    off += nelem * sizeof(_Float16);
    off = (off + 255) & ~(size_t)255;
    return p;
  };
  _Float16* xh  = alloc16((size_t)MALL * C);
  _Float16* wah = alloc16((size_t)N3 * C);
  _Float16* wph = alloc16((size_t)C * C);
  _Float16* Qh  = alloc16((size_t)BATCH * H * T * DH);
  _Float16* Kh  = alloc16((size_t)BATCH * H * T * DH);
  _Float16* Vth = alloc16((size_t)BATCH * H * T * DH);
  _Float16* yh  = alloc16((size_t)MALL * C);

  {
    int n = MALL * C;
    k_cvt_f16<<<(n + 255) / 256, 256, 0, stream>>>(x, xh, n);
  }
  {
    int n = C * N3;
    k_transpose_f16<<<(n + 255) / 256, 256, 0, stream>>>(w_attn, wah, C, N3);
  }
  {
    int n = C * C;
    k_transpose_f16<<<(n + 255) / 256, 256, 0, stream>>>(w_proj, wph, C, C);
  }
  // qkv = x @ w_attn + b_attn  -> Q/K/Vt f16
  k_gemm<<<dim3(N3 / 128, MALL / 128), 256, 0, stream>>>(
      xh, wah, b_attn, C, 0, Qh, Kh, Vth, nullptr, N3);
  // flash attention
  k_flash<<<BATCH * H * (T / 16), 32, 0, stream>>>(Qh, Kh, Vth, yh);
  // out = y @ w_proj + b_proj (f32)
  k_gemm<<<dim3(C / 128, MALL / 128), 256, 0, stream>>>(
      yh, wph, b_proj, C, 1, nullptr, nullptr, nullptr, out, C);
}
